// WeightedSumGraphRepresentation_51084341019407
// MI455X (gfx1250) — compile-verified, matrix operand order checked by reading, not run on hardware
//
#include <hip/hip_runtime.h>
#include <hip/hip_bf16.h>
#include <math.h>

typedef _Float16 f16;
typedef __attribute__((ext_vector_type(16))) _Float16 v16h;
typedef __attribute__((ext_vector_type(8)))  _Float16 v8h;
typedef __attribute__((ext_vector_type(4)))  _Float16 v4h;
typedef __attribute__((ext_vector_type(8)))  float    v8f;

#define IN_DIM   832
#define HID      128
#define NH       16
#define GD       416
#define NG       1024
#define MTILE    64          // rows per block (4 row-tiles of 16)
#define NEG_SLOPE 0.01f

__device__ __forceinline__ float lrelu(float v) { return v >= 0.f ? v : NEG_SLOPE * v; }

// Load a 16x32 f16 WMMA A/B fragment. Per CDNA5 ISA layout (wave32):
// lane holds row/col = lane&15; elements 0..7  = K in [base + 8*(lane>>4), +8)
//                      elements 8..15 = K in [base + 8*(lane>>4) + 16, +8)
// p must already point at (row * ldk + kbase + 8*(lane>>4)).
__device__ __forceinline__ v16h load_frag(const f16* p) {
  v8h lo = *(const v8h*)(p);
  v8h hi = *(const v8h*)(p + 16);
  v16h r;
#pragma unroll
  for (int i = 0; i < 8; ++i) { r[i] = lo[i]; r[i + 8] = hi[i]; }
  return r;
}

// ---------------- weight transpose + f32->f16 convert ----------------
// src: [K, OUT] row-major fp32 ; dst: [OUT, K] row-major f16
__global__ void k_transpose_f16(const float* __restrict__ src, f16* __restrict__ dst,
                                int K, int OUT) {
  int i = blockIdx.x * blockDim.x + threadIdx.x;
  if (i >= K * OUT) return;
  int j = i / K;      // output column (row of dst)
  int k = i - j * K;
  dst[(size_t)j * K + k] = (f16)src[(size_t)k * OUT + j];
}

// ---------------- fused node MLP (both branches) ----------------
// grid.x = N/64 ; block = 512 threads (16 waves)
// Each wave owns a 16-col tile and 4 row-tiles (4 f32 accumulators),
// so every B (weight) fragment is reused by 4 WMMAs.
__global__ __launch_bounds__(512)
void k_node_mlp(const float* __restrict__ x,
                const f16* __restrict__ wt0,   // [256][832] rows: 0-127 sW0^T, 128-255 tW0^T
                const f16* __restrict__ w1t,   // [256][128] rows: 0-127 sW1^T, 128-255 tW1^T
                const f16* __restrict__ w2s,   // [16][128]  sW2^T
                const f16* __restrict__ w2t,   // [416][128] tW2^T
                const float* __restrict__ b0,  // [256] {sb0, tb0}
                const float* __restrict__ b1,  // [256] {sb1, tb1}
                const float* __restrict__ b2s, // [16]
                const float* __restrict__ b2t, // [416]
                float* __restrict__ scores,    // [N][16]
                f16* __restrict__ nr)          // [N][416]
{
  // h2 overlays the x tile: x is dead after GEMM1, h2 is produced in GEMM2.
  __shared__ f16 sXH2[MTILE * IN_DIM];   // 104 KB
  __shared__ f16 sH[MTILE * 256];        // 32 KB
  f16* sX  = sXH2;
  f16* sH2 = sXH2;

  const int row0 = blockIdx.x * MTILE;
  const int wv   = threadIdx.x >> 5;
  const int lane = threadIdx.x & 31;
  const int ln16 = lane & 15;
  const int hi8  = (lane >> 4) * 8;

  // ---- cooperative x tile load (f32 -> f16), float4 loads, b64 LDS stores ----
  {
    const float4* xp = (const float4*)(x + (size_t)row0 * IN_DIM);
    v4h* dst = (v4h*)sX;
    for (int i = threadIdx.x; i < MTILE * IN_DIM / 4; i += 512) {
      float4 v = xp[i];
      v4h h;
      h[0] = (f16)v.x; h[1] = (f16)v.y; h[2] = (f16)v.z; h[3] = (f16)v.w;
      dst[i] = h;
    }
  }
  __syncthreads();

  // ---- GEMM1: [64,832] x [832,256] -> leaky -> sH ----
  {
    const int cbase = wv * 16;
    v8f acc[4];
    {
      const float bias = b0[cbase + ln16];
#pragma unroll
      for (int r = 0; r < 4; ++r)
#pragma unroll
        for (int i = 0; i < 8; ++i) acc[r][i] = bias;
    }
    const f16* bptr = wt0 + (size_t)(cbase + ln16) * IN_DIM + hi8;
    const f16* aptr = sX + ln16 * IN_DIM + hi8;
    for (int kk = 0; kk < IN_DIM; kk += 32) {
      v16h bf = load_frag(bptr + kk);
#pragma unroll
      for (int r = 0; r < 4; ++r) {
        v16h af = load_frag(aptr + r * 16 * IN_DIM + kk);
        acc[r] = __builtin_amdgcn_wmma_f32_16x16x32_f16(false, af, false, bf, (short)0,
                                                        acc[r], false, false);
      }
    }
#pragma unroll
    for (int r = 0; r < 4; ++r)
#pragma unroll
      for (int e = 0; e < 8; ++e)
        sH[(r * 16 + e + hi8) * 256 + cbase + ln16] = (f16)lrelu(acc[r][e]);
  }
  __syncthreads();

  // ---- GEMM2: s-branch (waves 0-7) / t-branch (waves 8-15), K=128 -> leaky -> sH2 ----
  {
    const int cbase = wv * 16;                 // col in [0,256)
    const int abase = (wv < 8) ? 0 : 128;      // which half of sH feeds this branch
    v8f acc[4];
    {
      const float bias = b1[cbase + ln16];
#pragma unroll
      for (int r = 0; r < 4; ++r)
#pragma unroll
        for (int i = 0; i < 8; ++i) acc[r][i] = bias;
    }
    const f16* bptr = w1t + (size_t)(cbase + ln16) * HID + hi8;
    const f16* aptr = sH + ln16 * 256 + abase + hi8;
#pragma unroll
    for (int kk = 0; kk < HID; kk += 32) {
      v16h bf = load_frag(bptr + kk);
#pragma unroll
      for (int r = 0; r < 4; ++r) {
        v16h af = load_frag(aptr + r * 16 * 256 + kk);
        acc[r] = __builtin_amdgcn_wmma_f32_16x16x32_f16(false, af, false, bf, (short)0,
                                                        acc[r], false, false);
      }
    }
    __syncthreads();   // sH2 overlays sX (already dead); also orders sH reads
#pragma unroll
    for (int r = 0; r < 4; ++r)
#pragma unroll
      for (int e = 0; e < 8; ++e)
        sH2[(r * 16 + e + hi8) * 256 + cbase + ln16] = (f16)lrelu(acc[r][e]);
  }
  __syncthreads();

  // ---- GEMM3: 26 repr col-tiles (t-branch) + 1 score tile (s-branch), x4 row tiles ----
  for (int T = wv; T < 27; T += 16) {
    if (T == 26) {
      // scores = h2_s @ sW2 + b2s  (no activation)
      v8f acc[4];
      {
        const float bias = b2s[ln16];
#pragma unroll
        for (int r = 0; r < 4; ++r)
#pragma unroll
          for (int i = 0; i < 8; ++i) acc[r][i] = bias;
      }
      const f16* bptr = w2s + (size_t)ln16 * HID + hi8;
      const f16* aptr = sH2 + ln16 * 256 + 0 + hi8;
#pragma unroll
      for (int kk = 0; kk < HID; kk += 32) {
        v16h bf = load_frag(bptr + kk);
#pragma unroll
        for (int r = 0; r < 4; ++r) {
          v16h af = load_frag(aptr + r * 16 * 256 + kk);
          acc[r] = __builtin_amdgcn_wmma_f32_16x16x32_f16(false, af, false, bf, (short)0,
                                                          acc[r], false, false);
        }
      }
#pragma unroll
      for (int r = 0; r < 4; ++r)
#pragma unroll
        for (int e = 0; e < 8; ++e)
          scores[(size_t)(row0 + r * 16 + e + hi8) * NH + ln16] = acc[r][e];
    } else {
      // node_reprs = leaky(h2_t @ tW2 + b2t), cols [16T,16T+16)
      const int cb = T * 16;
      v8f acc[4];
      {
        const float bias = b2t[cb + ln16];
#pragma unroll
        for (int r = 0; r < 4; ++r)
#pragma unroll
          for (int i = 0; i < 8; ++i) acc[r][i] = bias;
      }
      const f16* bptr = w2t + (size_t)(cb + ln16) * HID + hi8;
      const f16* aptr = sH2 + ln16 * 256 + 128 + hi8;
#pragma unroll
      for (int kk = 0; kk < HID; kk += 32) {
        v16h bf = load_frag(bptr + kk);
#pragma unroll
        for (int r = 0; r < 4; ++r) {
          v16h af = load_frag(aptr + r * 16 * 256 + kk);
          acc[r] = __builtin_amdgcn_wmma_f32_16x16x32_f16(false, af, false, bf, (short)0,
                                                          acc[r], false, false);
        }
      }
#pragma unroll
      for (int r = 0; r < 4; ++r)
#pragma unroll
        for (int e = 0; e < 8; ++e)
          nr[(size_t)(row0 + r * 16 + e + hi8) * GD + cb + ln16] = (f16)lrelu(acc[r][e]);
    }
  }
}

// ---------------- segment softmax + weighted segment-sum ----------------
// grid.x = NUM_GRAPHS ; block = 416 threads (13 waves)
__global__ __launch_bounds__(416)
void k_segment(const float* __restrict__ scores, const f16* __restrict__ nr,
               const int* __restrict__ batch, float* __restrict__ out, int N) {
  const int g = blockIdx.x;

  // binary search segment bounds in sorted batch[]
  int lo = 0, hi = N;
  while (lo < hi) { int mid = (lo + hi) >> 1; if (batch[mid] < g) lo = mid + 1; else hi = mid; }
  const int start = lo;
  hi = N;
  while (lo < hi) { int mid = (lo + hi) >> 1; if (batch[mid] < g + 1) lo = mid + 1; else hi = mid; }
  const int end = lo;

  __shared__ float red[26 * 16];
  __shared__ float s_mx[16];
  __shared__ float s_rinv[16];

  const int t    = threadIdx.x;   // 0..415
  const int head = t & 15;
  const int idx  = t >> 4;        // 0..25

  // per-head max
  float m = -INFINITY;
  for (int n = start + idx; n < end; n += 26)
    m = fmaxf(m, scores[(size_t)n * NH + head]);
  red[idx * 16 + head] = m;
  __syncthreads();
  if (t < 16) {
    float mm = -INFINITY;
#pragma unroll
    for (int i = 0; i < 26; ++i) mm = fmaxf(mm, red[i * 16 + t]);
    s_mx[t] = mm;
  }
  __syncthreads();

  // per-head sum of exp(score - max)
  const float mh = s_mx[head];
  float s = 0.f;
  for (int n = start + idx; n < end; n += 26)
    s += __expf(scores[(size_t)n * NH + head] - mh);
  red[idx * 16 + head] = s;
  __syncthreads();
  if (t < 16) {
    float ss = 0.f;
#pragma unroll
    for (int i = 0; i < 26; ++i) ss += red[i * 16 + t];
    s_rinv[t] = ss > 0.f ? 1.f / ss : 0.f;
  }
  __syncthreads();

  // weighted accumulation: thread owns output column c = t
  const int c = t;
  const int h = c / 26;
  const float mx = s_mx[h], rinv = s_rinv[h];
  float acc = 0.f;
  for (int n = start; n < end; ++n) {
    float w = __expf(scores[(size_t)n * NH + h] - mx) * rinv;
    acc += w * (float)nr[(size_t)n * GD + c];
  }
  out[(size_t)g * GD + c] = acc;
}

// ---------------- workspace layout (bytes, 256-aligned) ----------------
#define OFF_WT0    ((size_t)0)                        // f16 [256][832] = 425984
#define OFF_W1T    ((size_t)425984)                   // f16 [256][128] = 65536
#define OFF_W2S    ((size_t)491520)                   // f16 [16][128]  = 4096
#define OFF_W2T    ((size_t)495616)                   // f16 [416][128] = 106496
#define OFF_B0     ((size_t)602112)                   // f32 [256]
#define OFF_B1     ((size_t)603136)                   // f32 [256]
#define OFF_B2S    ((size_t)604160)                   // f32 [16]
#define OFF_B2T    ((size_t)604416)                   // f32 [416]
#define OFF_SCORES ((size_t)606208)                   // f32 [N][16]   = 8388608
#define OFF_NR     ((size_t)8994816)                  // f16 [N][416]  = 109051904

extern "C" void kernel_launch(void* const* d_in, const int* in_sizes, int n_in,
                              void* d_out, int out_size, void* d_ws, size_t ws_size,
                              hipStream_t stream) {
  const float* x    = (const float*)d_in[0];
  const int*   batch= (const int*)  d_in[1];
  const float* sW0  = (const float*)d_in[2];
  const float* sb0  = (const float*)d_in[3];
  const float* sW1  = (const float*)d_in[4];
  const float* sb1  = (const float*)d_in[5];
  const float* sW2  = (const float*)d_in[6];
  const float* sb2  = (const float*)d_in[7];
  const float* tW0  = (const float*)d_in[8];
  const float* tb0  = (const float*)d_in[9];
  const float* tW1  = (const float*)d_in[10];
  const float* tb1  = (const float*)d_in[11];
  const float* tW2  = (const float*)d_in[12];
  const float* tb2  = (const float*)d_in[13];
  float* out = (float*)d_out;

  const int N = in_sizes[0] / IN_DIM;   // 131072

  char* ws = (char*)d_ws;
  f16*   wt0    = (f16*)  (ws + OFF_WT0);
  f16*   w1t    = (f16*)  (ws + OFF_W1T);
  f16*   w2s    = (f16*)  (ws + OFF_W2S);
  f16*   w2t    = (f16*)  (ws + OFF_W2T);
  float* b0     = (float*)(ws + OFF_B0);
  float* b1     = (float*)(ws + OFF_B1);
  float* b2s    = (float*)(ws + OFF_B2S);
  float* b2t    = (float*)(ws + OFF_B2T);
  float* scores = (float*)(ws + OFF_SCORES);
  f16*   nr     = (f16*)  (ws + OFF_NR);

  // --- stage 0: weight convert/transpose + bias concat ---
  {
    dim3 blk(256);
    int n;
    n = IN_DIM * HID;  // 832*128
    k_transpose_f16<<<(n + 255) / 256, blk, 0, stream>>>(sW0, wt0,                 IN_DIM, HID);
    k_transpose_f16<<<(n + 255) / 256, blk, 0, stream>>>(tW0, wt0 + (size_t)HID * IN_DIM, IN_DIM, HID);
    n = HID * HID;
    k_transpose_f16<<<(n + 255) / 256, blk, 0, stream>>>(sW1, w1t,                 HID, HID);
    k_transpose_f16<<<(n + 255) / 256, blk, 0, stream>>>(tW1, w1t + (size_t)HID * HID,    HID, HID);
    n = HID * NH;
    k_transpose_f16<<<(n + 255) / 256, blk, 0, stream>>>(sW2, w2s, HID, NH);
    n = HID * GD;
    k_transpose_f16<<<(n + 255) / 256, blk, 0, stream>>>(tW2, w2t, HID, GD);

    hipMemcpyAsync(b0,        sb0, HID * sizeof(float), hipMemcpyDeviceToDevice, stream);
    hipMemcpyAsync(b0 + HID,  tb0, HID * sizeof(float), hipMemcpyDeviceToDevice, stream);
    hipMemcpyAsync(b1,        sb1, HID * sizeof(float), hipMemcpyDeviceToDevice, stream);
    hipMemcpyAsync(b1 + HID,  tb1, HID * sizeof(float), hipMemcpyDeviceToDevice, stream);
    hipMemcpyAsync(b2s,       sb2, NH  * sizeof(float), hipMemcpyDeviceToDevice, stream);
    hipMemcpyAsync(b2t,       tb2, GD  * sizeof(float), hipMemcpyDeviceToDevice, stream);
  }

  // --- stage 1: fused node MLPs (WMMA) ---
  k_node_mlp<<<N / MTILE, 512, 0, stream>>>(x, wt0, w1t, w2s, w2t, b0, b1, b2s, b2t,
                                            scores, nr);

  // --- stage 2: segment softmax + weighted segment sum ---
  k_segment<<<NG, 416, 0, stream>>>(scores, nr, batch, out, N);
}
